// ResidualBlock_42090679501106
// MI455X (gfx1250) — matrix operand order repaired and checked by reading, not compiled, not run on hardware
//
#include <hip/hip_runtime.h>
#include <hip/hip_bf16.h>

// ---------------------------------------------------------------------------
// KPConv residual block for MI455X (gfx1250, wave32, WMMA f16 16x16x32)
// ---------------------------------------------------------------------------

#define NPTS   65536
#define HNEI   32
#define CIN    256
#define CMID   64
#define COUT   256
#define KPK    15
#define SLOPE  0.1f
#define GNEPS  1e-5f

typedef __attribute__((ext_vector_type(16))) _Float16 v16h;
typedef __attribute__((ext_vector_type(8)))  _Float16 v8h;
typedef __attribute__((ext_vector_type(4)))  _Float16 v4h;
typedef __attribute__((ext_vector_type(8)))  float    v8f;
typedef __attribute__((ext_vector_type(4)))  int      v4i;

// ---------------------------------------------------------------------------
// WMMA fragment helpers (CDNA5 layouts, CDNA5 ISA §7.12.2)
// A (f16, 16x32): lane L holds row r=L&15; halves 0..7 = K[hi*8 .. hi*8+7],
//                 halves 8..15 = K[16+hi*8 .. 16+hi*8+7], hi = L>>4.
// B (f16, 32x16): lane L holds col c=L&15; halves j = K[(L>>4)*8 + (j&7) + 16*(j>>3)] (same as A).
//   -> pre-swizzled in global so each lane reads 16 contiguous halves.
// C (f32, 16x16): vgpr v, lane L -> row = v + 8*(L>>4), col = L&15.
// ---------------------------------------------------------------------------

__device__ __forceinline__ v16h load_A_f16(const _Float16* tile, int strideH,
                                           int kbase, int lane) {
  int r  = lane & 15;
  int hi = lane >> 4;
  const _Float16* p = tile + (size_t)r * strideH + kbase + hi * 8;
  v8h lo  = *(const v8h*)(p);
  v8h hi8 = *(const v8h*)(p + 16);
  return __builtin_shufflevector(lo, hi8, 0, 1, 2, 3, 4, 5, 6, 7,
                                 8, 9, 10, 11, 12, 13, 14, 15);
}

// swizzled-B element builder: idx -> (chunk q, dtile t, lane, j)
__device__ __forceinline__ void build_swz(const float* __restrict__ src,
                                          _Float16* __restrict__ dst,
                                          int Nd, int idx) {
  int j    = idx & 15;
  int lane = (idx >> 4) & 31;
  int qt   = idx >> 9;
  int numT = Nd >> 4;
  int t    = qt % numT;
  int q    = qt / numT;
  int row  = q * 32 + ((lane >> 4) << 3) + (j & 7) + ((j >> 3) << 4);   // K index
  int col  = (t << 4) + (lane & 15);            // N index
  dst[idx] = (_Float16)src[(size_t)row * Nd + col];
}

__device__ __forceinline__ float lrelu(float v) {
  return v >= 0.f ? v : SLOPE * v;
}

// ---------------------------------------------------------------------------
// K0: prep — swizzle W1 (256x64), kp_weights (960x64), W2 (64x256) to f16
//     B-fragment order; zero the GN stat accumulators.
// ---------------------------------------------------------------------------
#define NW1H  16384   // 8 chunks * 4 dtiles * 512
#define NKPH  61440   // 30 chunks * 4 dtiles * 512
#define NW2H  16384   // 2 chunks * 16 dtiles * 512
#define NSTAT 768     // statsA(128) statsB(128) statsC(512)

__global__ void prep_kernel(const float* __restrict__ W1,
                            const float* __restrict__ kpW,
                            const float* __restrict__ W2,
                            _Float16* __restrict__ w1h,
                            _Float16* __restrict__ kph,
                            _Float16* __restrict__ w2h,
                            float* __restrict__ stats) {
  int total = NSTAT + NW1H + NKPH + NW2H;
  for (int i = blockIdx.x * blockDim.x + threadIdx.x; i < total;
       i += gridDim.x * blockDim.x) {
    if (i < NSTAT) {
      stats[i] = 0.f;
    } else if (i < NSTAT + NW1H) {
      build_swz(W1, w1h, CMID, i - NSTAT);
    } else if (i < NSTAT + NW1H + NKPH) {
      build_swz(kpW, kph, CMID, i - NSTAT - NW1H);
    } else {
      build_swz(W2, w2h, COUT, i - NSTAT - NW1H - NKPH);
    }
  }
}

// ---------------------------------------------------------------------------
// K1: GEMM1  y1 = s_feats @ W1 + b1   (65536x256 @ 256x64, f32 out)
//     256 threads (8 waves), 64-row tile. Wave w: dtile w&3,
//     row-tiles {2*(w>>2), 2*(w>>2)+1}. Each B fragment feeds 2 WMMAs.
// ---------------------------------------------------------------------------
__global__ void gemm1_kernel(const float* __restrict__ A,
                             const _Float16* __restrict__ w1h,
                             const float* __restrict__ b1,
                             float* __restrict__ y1) {
  __shared__ __align__(16) _Float16 As[64 * 256];   // 32 KB
  int tid  = threadIdx.x;
  int row0 = blockIdx.x * 64;
  for (int i = tid; i < 64 * 256; i += 256) {
    As[i] = (_Float16)A[(size_t)(row0 + (i >> 8)) * 256 + (i & 255)];
  }
  __syncthreads();

  int lane = tid & 31;
  int wave = tid >> 5;
  int dt   = wave & 3;
  int rt0  = (wave >> 2) * 2;
  v8f acc0 = {0.f, 0.f, 0.f, 0.f, 0.f, 0.f, 0.f, 0.f};
  v8f acc1 = acc0;
#pragma unroll
  for (int q = 0; q < 8; ++q) {
    v16h b  = *(const v16h*)(w1h + ((size_t)(q * 4 + dt) * 32 + lane) * 16);
    v16h a0 = load_A_f16(As + (size_t)(rt0 * 16) * 256, 256, q * 32, lane);
    v16h a1 = load_A_f16(As + (size_t)(rt0 * 16 + 16) * 256, 256, q * 32, lane);
    acc0 = __builtin_amdgcn_wmma_f32_16x16x32_f16(false, a0, false, b,
                                                  (short)0, acc0, false, false);
    acc1 = __builtin_amdgcn_wmma_f32_16x16x32_f16(false, a1, false, b,
                                                  (short)0, acc1, false, false);
  }
  int col  = dt * 16 + (lane & 15);
  int hi   = lane >> 4;
  float bs = b1[col];
#pragma unroll
  for (int v = 0; v < 8; ++v) {
    int r = v + 8 * hi;
    y1[(size_t)(row0 + rt0 * 16 + r) * CMID + col]        = acc0[v] + bs;
    y1[(size_t)(row0 + rt0 * 16 + 16 + r) * CMID + col]   = acc1[v] + bs;
  }
}

// ---------------------------------------------------------------------------
// K2/K5/K8: per-channel sum/sumsq reduction (GroupNorm stats over all N).
//     gridDim*blockDim is a multiple of C -> each thread's channel is fixed.
// ---------------------------------------------------------------------------
__global__ void stats_kernel(const float* __restrict__ src, int C,
                             float* __restrict__ accum, size_t total) {
  __shared__ float sh[512];
  int tid = threadIdx.x;
  for (int i = tid; i < 2 * C; i += blockDim.x) sh[i] = 0.f;
  __syncthreads();
  size_t start  = (size_t)blockIdx.x * blockDim.x + tid;
  size_t stride = (size_t)gridDim.x * blockDim.x;
  int c = (int)(start % (size_t)C);
  float s = 0.f, ss = 0.f;
  for (size_t i = start; i < total; i += stride) {
    float v = src[i];
    s += v;
    ss += v * v;
  }
  atomicAdd(&sh[c], s);
  atomicAdd(&sh[C + c], ss);
  __syncthreads();
  for (int i = tid; i < 2 * C; i += blockDim.x) atomicAdd(&accum[i], sh[i]);
}

// ---------------------------------------------------------------------------
// K3/K6: apply GroupNorm (+LeakyReLU) and cast to f16. totalPad > total
//     zero-fills the shadow row used by the neighbor gather.
// ---------------------------------------------------------------------------
__global__ void gn_apply_half(const float* __restrict__ y,
                              const float* __restrict__ st,
                              const float* __restrict__ gamma,
                              const float* __restrict__ beta,
                              _Float16* __restrict__ dst,
                              int C, int chPerG, size_t total, size_t totalPad) {
  size_t start  = (size_t)blockIdx.x * blockDim.x + threadIdx.x;
  size_t stride = (size_t)gridDim.x * blockDim.x;
  for (size_t i = start; i < totalPad; i += stride) {
    if (i >= total) { dst[i] = (_Float16)0.f; continue; }
    int c = (int)(i % (size_t)C);
    int g = c / chPerG;
    float sm = 0.f, sq = 0.f;
    for (int cc = g * chPerG; cc < (g + 1) * chPerG; ++cc) {
      sm += st[cc];
      sq += st[C + cc];
    }
    float cnt  = (float)NPTS * (float)chPerG;
    float mean = sm / cnt;
    float var  = sq / cnt - mean * mean;
    float rstd = rsqrtf(var + GNEPS);
    float v = (y[i] - mean) * rstd * gamma[c] + beta[c];
    dst[i] = (_Float16)lrelu(v);
  }
}

// ---------------------------------------------------------------------------
// K4: KPConv. 256 threads (8 waves), 16 query points / block.
//   Phase 0: geometry weights w[p][h][k] in LDS.
//   Phase 1: M[p] = w^T * gathered_feats  (register accumulators, f32).
//   Phase 2: out = M(16x960) @ Wflat(960x64) via 30 WMMA K-chunks,
//            split across wave pairs (waves 0-3 chunks 0..14, 4-7 15..29).
// ---------------------------------------------------------------------------
__global__ void kpconv_kernel(const _Float16* __restrict__ x1h,
                              const int* __restrict__ nidx_g,
                              const float* __restrict__ qpts,
                              const float* __restrict__ spts,
                              const float* __restrict__ kpts,
                              const _Float16* __restrict__ kph,
                              const float* __restrict__ kpb,
                              float* __restrict__ y2) {
  __shared__ float wl[16][32][16];                  // 32 KB (k padded to 16)
  __shared__ __align__(16) _Float16 Ml[16][960];    // 30 KB
  __shared__ int   ni[16][32];
  __shared__ float kp[15][3];
  __shared__ float qp[16][3];
  __shared__ int   nn[16];
  __shared__ float red[4][256];                     // cross-wave partials

  int tid  = threadIdx.x;
  int row0 = blockIdx.x * 16;

  for (int i = tid; i < 512; i += 256)
    ni[i >> 5][i & 31] = nidx_g[(size_t)row0 * HNEI + i];
  if (tid < 45) ((float*)kp)[tid] = kpts[tid];
  if (tid < 48) ((float*)qp)[tid] = qpts[(size_t)row0 * 3 + tid];
  __syncthreads();

  // --- phase 0: kernel-point influence weights -----------------------------
  for (int pr = tid; pr < 512; pr += 256) {
    int p = pr >> 5, h = pr & 31;
    int idx = ni[p][h];
    float sx, sy, sz;
    if (idx < NPTS) {
      sx = spts[(size_t)idx * 3 + 0];
      sy = spts[(size_t)idx * 3 + 1];
      sz = spts[(size_t)idx * 3 + 2];
    } else {
      sx = sy = sz = 1e10f;   // shadow point -> w = 0
    }
    float dx = sx - qp[p][0], dy = sy - qp[p][1], dz = sz - qp[p][2];
#pragma unroll
    for (int k = 0; k < KPK; ++k) {
      float ex = dx - kp[k][0], ey = dy - kp[k][1], ez = dz - kp[k][2];
      float dist = sqrtf(ex * ex + ey * ey + ez * ez);
      float wv = 1.f - dist;                         // SIGMA = 1
      wl[p][h][k] = wv > 0.f ? wv : 0.f;
    }
  }
  if (tid < 16) {
    int c = 0;
    for (int h = 0; h < HNEI; ++h) c += (ni[tid][h] < NPTS) ? 1 : 0;
    nn[tid] = c > 0 ? c : 1;
  }
  __syncthreads();

  // --- phase 1: M[p][k][c] = sum_h w[p][h][k] * nf[p][h][c] ---------------
  {
    int p  = tid >> 4;
    int c0 = (tid & 15) * 4;
    float acc[KPK][4];
#pragma unroll
    for (int k = 0; k < KPK; ++k)
#pragma unroll
      for (int j = 0; j < 4; ++j) acc[k][j] = 0.f;

    for (int h = 0; h < HNEI; ++h) {
      int idx = ni[p][h];
      // prefetch next gather row (global_prefetch_b8)
      int idxn = ni[p][(h + 1) & 31];
      __builtin_prefetch(x1h + (size_t)idxn * CMID, 0, 0);
      v4h nf = *(const v4h*)(x1h + (size_t)idx * CMID + c0);
      float f0 = (float)nf[0], f1 = (float)nf[1];
      float f2 = (float)nf[2], f3 = (float)nf[3];
#pragma unroll
      for (int k = 0; k < KPK; ++k) {
        float wv = wl[p][h][k];
        acc[k][0] += wv * f0;
        acc[k][1] += wv * f1;
        acc[k][2] += wv * f2;
        acc[k][3] += wv * f3;
      }
    }
#pragma unroll
    for (int k = 0; k < KPK; ++k)
#pragma unroll
      for (int j = 0; j < 4; ++j)
        Ml[p][k * CMID + c0 + j] = (_Float16)acc[k][j];
  }
  __syncthreads();

  // --- phase 2: WMMA GEMM, 16x960 @ 960x64 --------------------------------
  int lane = tid & 31;
  int wave = tid >> 5;
  int dt   = wave & 3;
  int hlf  = wave >> 2;
  v8f o = {0.f, 0.f, 0.f, 0.f, 0.f, 0.f, 0.f, 0.f};
#pragma unroll
  for (int qq = 0; qq < 15; ++qq) {
    int q = hlf * 15 + qq;
    v16h a = load_A_f16(&Ml[0][0], 960, q * 32, lane);
    v16h b = *(const v16h*)(kph + ((size_t)(q * 4 + dt) * 32 + lane) * 16);
    o = __builtin_amdgcn_wmma_f32_16x16x32_f16(false, a, false, b,
                                               (short)0, o, false, false);
  }
  if (hlf == 1) {
#pragma unroll
    for (int v = 0; v < 8; ++v) red[dt][v * 32 + lane] = o[v];
  }
  __syncthreads();
  if (hlf == 0) {
    int col = dt * 16 + (lane & 15);
    int hi  = lane >> 4;
    float bs = kpb[col];
#pragma unroll
    for (int v = 0; v < 8; ++v) {
      int r = v + 8 * hi;
      float val = (o[v] + red[dt][v * 32 + lane]) / (float)nn[r] + bs;
      y2[(size_t)(row0 + r) * CMID + col] = val;
    }
  }
}

// ---------------------------------------------------------------------------
// K7: GEMM2  y3 = x2 @ W2 + b2   (65536x64 @ 64x256, f32 out)
//     256 threads (8 waves), 64-row tile. Wave w: dtiles {w, w+8},
//     all 4 row-tiles -> each B fragment feeds 4 WMMAs.
//     A tile staged with GLOBAL_LOAD_ASYNC_TO_LDS when the builtin exists.
// ---------------------------------------------------------------------------
__global__ void gemm2_kernel(const _Float16* __restrict__ x2h,
                             const _Float16* __restrict__ w2h,
                             const float* __restrict__ b2,
                             float* __restrict__ y3) {
  __shared__ __align__(16) _Float16 As[64 * 64];    // 8 KB
  int tid  = threadIdx.x;
  int row0 = blockIdx.x * 64;

#if __has_builtin(__builtin_amdgcn_global_load_async_to_lds_b128)
  {
    // 64 rows * 128 B/row = 512 x b128 DMA transfers, ASYNCcnt-tracked.
#pragma unroll
    for (int it = 0; it < 2; ++it) {
      int e   = tid + it * 256;
      int row = e >> 3;
      int c8  = (e & 7) * 8;
      __builtin_amdgcn_global_load_async_to_lds_b128(
          (__attribute__((address_space(1))) v4i*)(
              x2h + (size_t)(row0 + row) * CMID + c8),
          (__attribute__((address_space(3))) v4i*)(As + row * 64 + c8),
          0, 0);
    }
#if __has_builtin(__builtin_amdgcn_s_wait_asynccnt)
    __builtin_amdgcn_s_wait_asynccnt(0);
#else
    asm volatile("s_wait_asynccnt 0x0" ::: "memory");
#endif
  }
#else
  for (int i = tid; i < 64 * 64; i += 256) {
    As[i] = x2h[(size_t)(row0 + (i >> 6)) * CMID + (i & 63)];
  }
#endif
  __syncthreads();

  int lane = tid & 31;
  int wave = tid >> 5;
  int hi   = lane >> 4;
#pragma unroll
  for (int s = 0; s < 2; ++s) {
    int dt = wave + s * 8;
    v8f acc[4];
#pragma unroll
    for (int rt = 0; rt < 4; ++rt)
      acc[rt] = (v8f){0.f, 0.f, 0.f, 0.f, 0.f, 0.f, 0.f, 0.f};
#pragma unroll
    for (int q = 0; q < 2; ++q) {
      v16h b = *(const v16h*)(w2h + ((size_t)(q * 16 + dt) * 32 + lane) * 16);
#pragma unroll
      for (int rt = 0; rt < 4; ++rt) {
        v16h a = load_A_f16(As + (size_t)(rt * 16) * 64, 64, q * 32, lane);
        acc[rt] = __builtin_amdgcn_wmma_f32_16x16x32_f16(
            false, a, false, b, (short)0, acc[rt], false, false);
      }
    }
    int col  = dt * 16 + (lane & 15);
    float bs = b2[col];
#pragma unroll
    for (int rt = 0; rt < 4; ++rt) {
#pragma unroll
      for (int v = 0; v < 8; ++v) {
        int r = v + 8 * hi;
        y3[(size_t)(row0 + rt * 16 + r) * COUT + col] = acc[rt][v] + bs;
      }
    }
  }
}

// ---------------------------------------------------------------------------
// K9: final GN + residual + LeakyReLU -> d_out (f32, N x 256)
// ---------------------------------------------------------------------------
__global__ void final_kernel(const float* __restrict__ y3,
                             const float* __restrict__ st,
                             const float* __restrict__ g2,
                             const float* __restrict__ beta2,
                             const float* __restrict__ sfeat,
                             float* __restrict__ out, size_t total) {
  size_t start  = (size_t)blockIdx.x * blockDim.x + threadIdx.x;
  size_t stride = (size_t)gridDim.x * blockDim.x;
  for (size_t i = start; i < total; i += stride) {
    int c = (int)(i & 255);
    int g = c >> 3;                      // 8 channels / group
    float sm = 0.f, sq = 0.f;
#pragma unroll
    for (int cc = g * 8; cc < g * 8 + 8; ++cc) {
      sm += st[cc];
      sq += st[COUT + cc];
    }
    float cnt  = (float)NPTS * 8.f;
    float mean = sm / cnt;
    float var  = sq / cnt - mean * mean;
    float rstd = rsqrtf(var + GNEPS);
    float v = (y3[i] - mean) * rstd * g2[c] + beta2[c] + sfeat[i];
    out[i] = lrelu(v);
  }
}

// ---------------------------------------------------------------------------
// Launch
// ---------------------------------------------------------------------------
static inline size_t align256(size_t x) { return (x + 255) & ~(size_t)255; }

extern "C" void kernel_launch(void* const* d_in, const int* in_sizes, int n_in,
                              void* d_out, int out_size, void* d_ws, size_t ws_size,
                              hipStream_t stream) {
  const float* s_feats  = (const float*)d_in[0];
  const float* q_points = (const float*)d_in[1];
  const float* s_points = (const float*)d_in[2];
  const int*   nidx     = (const int*)d_in[3];
  const float* W1       = (const float*)d_in[4];
  const float* b1       = (const float*)d_in[5];
  const float* g1       = (const float*)d_in[6];
  const float* beta1    = (const float*)d_in[7];
  const float* kpW      = (const float*)d_in[8];
  const float* kpb      = (const float*)d_in[9];
  const float* kpts     = (const float*)d_in[10];
  const float* gg       = (const float*)d_in[11];
  const float* gb       = (const float*)d_in[12];
  const float* W2       = (const float*)d_in[13];
  const float* b2       = (const float*)d_in[14];
  const float* g2       = (const float*)d_in[15];
  const float* beta2    = (const float*)d_in[16];
  float* out = (float*)d_out;
  char*  ws  = (char*)d_ws;

  size_t off = 0;
  float*    y1  = (float*)(ws + off);    off = align256(off + (size_t)NPTS * CMID * 4);
  _Float16* x1h = (_Float16*)(ws + off); off = align256(off + (size_t)(NPTS + 1) * CMID * 2);
  float*    y2  = (float*)(ws + off);    off = align256(off + (size_t)NPTS * CMID * 4);
  _Float16* x2h = (_Float16*)(ws + off); off = align256(off + (size_t)NPTS * CMID * 2);
  float*    y3  = (float*)(ws + off);    off = align256(off + (size_t)NPTS * COUT * 4);
  _Float16* w1h = (_Float16*)(ws + off); off = align256(off + (size_t)NW1H * 2);
  _Float16* kph = (_Float16*)(ws + off); off = align256(off + (size_t)NKPH * 2);
  _Float16* w2h = (_Float16*)(ws + off); off = align256(off + (size_t)NW2H * 2);
  float*    stats  = (float*)(ws + off);
  float*    statsA = stats;        // 128
  float*    statsB = stats + 128;  // 128
  float*    statsC = stats + 256;  // 512

  prep_kernel<<<256, 256, 0, stream>>>(W1, kpW, W2, w1h, kph, w2h, stats);

  gemm1_kernel<<<NPTS / 64, 256, 0, stream>>>(s_feats, w1h, b1, y1);
  stats_kernel<<<1024, 256, 0, stream>>>(y1, CMID, statsA, (size_t)NPTS * CMID);
  gn_apply_half<<<2048, 256, 0, stream>>>(y1, statsA, g1, beta1, x1h, CMID, 2,
                                          (size_t)NPTS * CMID,
                                          (size_t)(NPTS + 1) * CMID);

  kpconv_kernel<<<NPTS / 16, 256, 0, stream>>>(x1h, nidx, q_points, s_points,
                                               kpts, kph, kpb, y2);
  stats_kernel<<<1024, 256, 0, stream>>>(y2, CMID, statsB, (size_t)NPTS * CMID);
  gn_apply_half<<<2048, 256, 0, stream>>>(y2, statsB, gg, gb, x2h, CMID, 2,
                                          (size_t)NPTS * CMID,
                                          (size_t)NPTS * CMID);

  gemm2_kernel<<<NPTS / 64, 256, 0, stream>>>(x2h, w2h, b2, y3);
  stats_kernel<<<1024, 256, 0, stream>>>(y3, COUT, statsC, (size_t)NPTS * COUT);
  final_kernel<<<4096, 256, 0, stream>>>(y3, statsC, g2, beta2, s_feats, out,
                                         (size_t)NPTS * COUT);
}
